// Attention_73718818669284
// MI455X (gfx1250) — matrix operand order repaired and checked by reading, not compiled, not run on hardware
//
#include <hip/hip_runtime.h>

// ---------------------------------------------------------------------------
// Problem constants (from reference): B=2, C=128, N=16*16*16=4096, 8 groups.
// ---------------------------------------------------------------------------
#define BB 2
#define CC 128
#define NN 4096
#define NGRP 8
#define GN_EPS 1e-5f

typedef __attribute__((ext_vector_type(16))) __bf16 v16bf;
typedef __attribute__((ext_vector_type(8)))  __bf16 v8bf;
typedef __attribute__((ext_vector_type(8)))  float  v8f;

// ---------------------------------------------------------------------------
// Fragment helpers (CDNA5 wave32 WMMA layouts, 16x16x32 bf16 -> f32)
//
// A (16x32, row-major src, stride ld):
//   lane: row = lane&15, half = lane>>4
//   elems 0..7  = A[row][half*8 + 0..7]
//   elems 8..15 = A[row][16 + half*8 + 0..7]
// B (32x16, src stored [col][k] row-major, stride ld):
//   lane: col = lane&15, half = lane>>4
//   elems 0..7  = B[half*8 + 0..7][col], elems 8..15 = B[16 + half*8 + 0..7][col]
// C/D (16x16 f32): d[r] = D[half*8 + r][lane&15]
// ---------------------------------------------------------------------------
__device__ __forceinline__ v16bf load_frag_a(const __bf16* base, int row_base,
                                             int col_base, int ld) {
  int lane = threadIdx.x & 31;
  int half = lane >> 4, row = lane & 15;
  const __bf16* p = base + (size_t)(row_base + row) * ld + col_base + half * 8;
  v8bf lo = *(const v8bf*)p;
  v8bf hi = *(const v8bf*)(p + 16);
  return __builtin_shufflevector(lo, hi, 0, 1, 2, 3, 4, 5, 6, 7,
                                 8, 9, 10, 11, 12, 13, 14, 15);
}

__device__ __forceinline__ v16bf load_frag_b(const __bf16* base, int col_base,
                                             int k_base, int ld) {
  int lane = threadIdx.x & 31;
  int half = lane >> 4, col = lane & 15;
  const __bf16* p = base + (size_t)(col_base + col) * ld + k_base + half * 8;
  v8bf lo = *(const v8bf*)p;
  v8bf hi = *(const v8bf*)(p + 16);
  return __builtin_shufflevector(lo, hi, 0, 1, 2, 3, 4, 5, 6, 7,
                                 8, 9, 10, 11, 12, 13, 14, 15);
}

__device__ __forceinline__ v8f wmma_bf16(v16bf a, v16bf b, v8f c) {
  return __builtin_amdgcn_wmma_f32_16x16x32_bf16(
      /*neg_a=*/false, a, /*neg_b=*/false, b,
      /*c_mod=*/(short)0, c, /*reuse_a=*/false, /*reuse_b=*/false);
}

// ---------------------------------------------------------------------------
// Kernel 0: x[b][c][n] (f32) -> xt[b][n][c] (bf16), LDS-tiled transpose
// ---------------------------------------------------------------------------
__global__ void k_transpose(const float* __restrict__ x, __bf16* __restrict__ xt) {
  __shared__ float tile[16][17];
  int b = blockIdx.z;
  int n0 = blockIdx.x * 16, c0 = blockIdx.y * 16;
  const float* xb = x + (size_t)b * CC * NN;
  tile[threadIdx.y][threadIdx.x] = xb[(size_t)(c0 + threadIdx.y) * NN + n0 + threadIdx.x];
  __syncthreads();
  xt[((size_t)b * NN + n0 + threadIdx.y) * CC + c0 + threadIdx.x] =
      (__bf16)tile[threadIdx.x][threadIdx.y];
}

// ---------------------------------------------------------------------------
// Kernel 1: weight f32 -> bf16 (Wq | Wk | Wv | Wo concatenated, row-major)
// ---------------------------------------------------------------------------
__global__ void k_wconv(const float* __restrict__ Wq, const float* __restrict__ Wk,
                        const float* __restrict__ Wv, const float* __restrict__ Wo,
                        __bf16* __restrict__ wbf) {
  int i = blockIdx.x * 256 + threadIdx.x;           // < 4*128*128 = 65536
  int m = i >> 14, l = i & 16383;
  const float* src = (m == 0) ? Wq : (m == 1) ? Wk : (m == 2) ? Wv : Wo;
  wbf[i] = (__bf16)src[l];
}

// ---------------------------------------------------------------------------
// Kernel 2: fused QKV projection via WMMA.
//   Qt[n][o] = sum_c xt[n][c] * Wq[o][c]   (A=xt, B=Wq)
//   Kt[n][o] = sum_c xt[n][c] * Wk[o][c]
//   V [o][n] = sum_c Wv[o][c] * xt[n][c]   (A=Wv, B=xt)
// One wave per 16x16 output tile; K-loop = 4 WMMAs (C=128).
// ---------------------------------------------------------------------------
__global__ void k_proj(const __bf16* __restrict__ xt, const __bf16* __restrict__ wbf,
                       __bf16* __restrict__ qt, __bf16* __restrict__ kt,
                       __bf16* __restrict__ vb) {
  int gw = blockIdx.x * (blockDim.x >> 5) + (threadIdx.x >> 5);
  int mat = gw >> 12;            // 4096 tiles per matrix (2 batches x 2048)
  int rem = gw & 4095;
  int b = rem >> 11;
  int t = rem & 2047;            // 256 n-tiles x 8 o-tiles
  int ntile = t >> 3, ot = t & 7;
  int lane = threadIdx.x & 31, half = lane >> 4, lc = lane & 15;
  const __bf16* xb = xt + (size_t)b * NN * CC;
  v8f acc = {0.f, 0.f, 0.f, 0.f, 0.f, 0.f, 0.f, 0.f};

  if (mat < 2) {
    const __bf16* W = wbf + (size_t)mat * 16384;    // Wq or Wk
#pragma unroll
    for (int cc = 0; cc < 4; ++cc) {
      v16bf a = load_frag_a(xb, ntile * 16, cc * 32, CC);
      v16bf bf = load_frag_b(W, ot * 16, cc * 32, CC);
      acc = wmma_bf16(a, bf, acc);
    }
    __bf16* dst = ((mat == 0) ? qt : kt) + (size_t)b * NN * CC;
#pragma unroll
    for (int r = 0; r < 8; ++r)
      dst[(size_t)(ntile * 16 + half * 8 + r) * CC + ot * 16 + lc] = (__bf16)acc[r];
  } else {
    const __bf16* W = wbf + (size_t)2 * 16384;      // Wv
#pragma unroll
    for (int cc = 0; cc < 4; ++cc) {
      v16bf a = load_frag_a(W, ot * 16, cc * 32, CC);
      v16bf bf = load_frag_b(xb, ntile * 16, cc * 32, CC);
      acc = wmma_bf16(a, bf, acc);
    }
    __bf16* dst = vb + (size_t)b * CC * NN;
#pragma unroll
    for (int r = 0; r < 8; ++r)
      dst[(size_t)(ot * 16 + half * 8 + r) * NN + ntile * 16 + lc] = (__bf16)acc[r];
  }
}

// ---------------------------------------------------------------------------
// Async K/V tile prefetch into LDS (ASYNCcnt path, gfx1250).
// All 128 threads cooperate; 8 x global_load_async_to_lds_b128 per call:
//   K tile: Kt[m0..m0+32)[0..128) bf16 -> 8 KB contiguous
//   V tile: V[0..128)[m0..m0+32) bf16 -> 128 rows x 64 B (row stride 8 KB)
// ---------------------------------------------------------------------------
__device__ __forceinline__ void prefetch_kv(const __bf16* Kb, const __bf16* Vb,
                                            int m0, __bf16* kdst, __bf16* vdst,
                                            int tid) {
  uint64_t kbase = (uint64_t)(uintptr_t)Kb;
  uint32_t klds = (uint32_t)(uintptr_t)kdst;     // generic->LDS: low 32 bits
#pragma unroll
  for (int j = 0; j < 4; ++j) {
    uint32_t byte = (uint32_t)(j * 128 + tid) * 16u;
    uint32_t goff = (uint32_t)m0 * (CC * 2u) + byte;
    uint32_t ldsa = klds + byte;
    asm volatile("global_load_async_to_lds_b128 %0, %1, %2"
                 :: "v"(ldsa), "v"(goff), "s"(kbase) : "memory");
  }
  uint64_t vbase = (uint64_t)(uintptr_t)Vb;
  uint32_t vlds = (uint32_t)(uintptr_t)vdst;
#pragma unroll
  for (int j = 0; j < 4; ++j) {
    uint32_t row = (uint32_t)(j * 32 + (tid >> 2));
    uint32_t boff = (uint32_t)(tid & 3) * 16u;
    uint32_t goff = row * (NN * 2u) + (uint32_t)m0 * 2u + boff;
    uint32_t ldsa = vlds + row * 64u + boff;
    asm volatile("global_load_async_to_lds_b128 %0, %1, %2"
                 :: "v"(ldsa), "v"(goff), "s"(vbase) : "memory");
  }
}

// ---------------------------------------------------------------------------
// Kernel 3: flash attention, block = 4 waves sharing one K/V stream.
//   wave w owns query rows [ntile*16, ntile*16+16); all waves share batch b.
//   Double-buffered async K/V staging in LDS; WMMA S=Q K^T, online softmax,
//   P bounced through wave-private LDS to A-layout, WMMA O += P V^T.
// ---------------------------------------------------------------------------
__global__ void k_attn(const __bf16* __restrict__ qt, const __bf16* __restrict__ kt,
                       const __bf16* __restrict__ vb, __bf16* __restrict__ ht) {
  __shared__ __align__(16) __bf16 kbuf[2][32 * 128];   // [m][c], 8 KB each
  __shared__ __align__(16) __bf16 vbuf[2][128 * 32];   // [c][m], 8 KB each
  __shared__ __align__(16) __bf16 plds[4][16 * 32];    // per-wave P bounce

  int tid = threadIdx.x;
  int wave = tid >> 5;
  int lane = tid & 31;
  int half = lane >> 4, lc = lane & 15;
  int b = blockIdx.x >> 6;                       // 64 blocks per batch
  int ntile = (blockIdx.x & 63) * 4 + wave;      // 4 n-tiles per block
  const __bf16* Qb = qt + (size_t)b * NN * CC;
  const __bf16* Kb = kt + (size_t)b * NN * CC;
  const __bf16* Vb = vb + (size_t)b * CC * NN;

  v16bf aQ[4];
#pragma unroll
  for (int cc = 0; cc < 4; ++cc) aQ[cc] = load_frag_a(Qb, ntile * 16, cc * 32, CC);

  v8f o[8];
#pragma unroll
  for (int ct = 0; ct < 8; ++ct) o[ct] = (v8f){0.f, 0.f, 0.f, 0.f, 0.f, 0.f, 0.f, 0.f};
  float mrun[8], lrun[8];
#pragma unroll
  for (int r = 0; r < 8; ++r) { mrun[r] = -1e30f; lrun[r] = 0.f; }

  __bf16* pl = plds[wave];
  const int T = NN / 32;                         // 128 steps

  prefetch_kv(Kb, Vb, 0, kbuf[0], vbuf[0], tid);

  for (int mt = 0; mt < T; ++mt) {
    int cur = mt & 1;
    if (mt + 1 < T) {
      prefetch_kv(Kb, Vb, (mt + 1) * 32, kbuf[cur ^ 1], vbuf[cur ^ 1], tid);
      // 8 just-issued may remain; async loads complete in order, so <=8
      // guarantees the current buffer's 8 transfers have landed.
      asm volatile("s_wait_asynccnt 0x8" ::: "memory");
    } else {
      asm volatile("s_wait_asynccnt 0x0" ::: "memory");
    }
    __syncthreads();                             // all waves' portions visible

    const __bf16* kc = kbuf[cur];
    const __bf16* vc = vbuf[cur];
    v8f s0 = {0.f, 0.f, 0.f, 0.f, 0.f, 0.f, 0.f, 0.f};
    v8f s1 = {0.f, 0.f, 0.f, 0.f, 0.f, 0.f, 0.f, 0.f};
#pragma unroll
    for (int cc = 0; cc < 4; ++cc) {
      v16bf bk0 = load_frag_b(kc, 0, cc * 32, CC);
      s0 = wmma_bf16(aQ[cc], bk0, s0);
      v16bf bk1 = load_frag_b(kc, 16, cc * 32, CC);
      s1 = wmma_bf16(aQ[cc], bk1, s1);
    }
    // --- online softmax (rows live across the 16 lanes of each half-wave) ---
    float p0[8], p1[8];
#pragma unroll
    for (int r = 0; r < 8; ++r) {
      float mx = fmaxf(s0[r], s1[r]);
#pragma unroll
      for (int d = 1; d < 16; d <<= 1) mx = fmaxf(mx, __shfl_xor(mx, d, 32));
      float mnew = fmaxf(mrun[r], mx);
      float alpha = __expf(mrun[r] - mnew);
      float e0 = __expf(s0[r] - mnew);
      float e1 = __expf(s1[r] - mnew);
      float rs = e0 + e1;
#pragma unroll
      for (int d = 1; d < 16; d <<= 1) rs += __shfl_xor(rs, d, 32);
      lrun[r] = lrun[r] * alpha + rs;
      mrun[r] = mnew;
      p0[r] = e0; p1[r] = e1;
#pragma unroll
      for (int ct = 0; ct < 8; ++ct) o[ct][r] *= alpha;
    }
    // --- bounce P (D-layout) through wave-private LDS into A-layout ---
#pragma unroll
    for (int r = 0; r < 8; ++r) {
      pl[(half * 8 + r) * 32 + lc] = (__bf16)p0[r];
      pl[(half * 8 + r) * 32 + 16 + lc] = (__bf16)p1[r];
    }
    asm volatile("s_wait_dscnt 0x0" ::: "memory");  // wave-private: DS in-order
    v16bf aP = load_frag_a(pl, 0, 0, 32);
#pragma unroll
    for (int ct = 0; ct < 8; ++ct) {
      v16bf bv = load_frag_b(vc, ct * 16, 0, 32);
      o[ct] = wmma_bf16(aP, bv, o[ct]);
    }
    __syncthreads();   // everyone done with buffer `cur` before it is refilled
  }

  float inv[8];
#pragma unroll
  for (int r = 0; r < 8; ++r) inv[r] = 1.0f / lrun[r];
  __bf16* hb = ht + (size_t)b * NN * CC;
#pragma unroll
  for (int ct = 0; ct < 8; ++ct)
#pragma unroll
    for (int r = 0; r < 8; ++r)
      hb[(size_t)(ntile * 16 + half * 8 + r) * CC + ct * 16 + lc] =
          (__bf16)(o[ct][r] * inv[r]);
}

// ---------------------------------------------------------------------------
// Kernel 4: out projection + residual.  y[o][n] = x[o][n] + sum_c Wo[o][c] Ht[n][c]
// ---------------------------------------------------------------------------
__global__ void k_oproj(const __bf16* __restrict__ wbf, const __bf16* __restrict__ ht,
                        const float* __restrict__ x, float* __restrict__ y) {
  int gw = blockIdx.x * 4 + (threadIdx.x >> 5);    // 4096 waves
  int b = gw >> 11;
  int t = gw & 2047;
  int ntile = t >> 3, ot = t & 7;
  int lane = threadIdx.x & 31, half = lane >> 4, lc = lane & 15;
  const __bf16* Wo = wbf + (size_t)3 * 16384;
  const __bf16* hb = ht + (size_t)b * NN * CC;
  v8f acc = {0.f, 0.f, 0.f, 0.f, 0.f, 0.f, 0.f, 0.f};
#pragma unroll
  for (int cc = 0; cc < 4; ++cc) {
    v16bf a = load_frag_a(Wo, ot * 16, cc * 32, CC);
    v16bf bf = load_frag_b(hb, ntile * 16, cc * 32, CC);
    acc = wmma_bf16(a, bf, acc);
  }
  const float* xb = x + (size_t)b * CC * NN;
  float* yb = y + (size_t)b * CC * NN;
#pragma unroll
  for (int r = 0; r < 8; ++r) {
    size_t idx = (size_t)(ot * 16 + half * 8 + r) * NN + ntile * 16 + lc;
    yb[idx] = xb[idx] + acc[r];
  }
}

// ---------------------------------------------------------------------------
// Kernel 5: GroupNorm stats (one block per (b, group); 16*4096 elems each)
// ---------------------------------------------------------------------------
__global__ void k_gnstats(const float* __restrict__ y, float* __restrict__ stats) {
  __shared__ float ssum[256], ssq[256];
  int bg = blockIdx.x;                              // 0..15
  const float* p = y + (size_t)bg * (16 * NN);      // groups are channel-contiguous
  float s = 0.f, q = 0.f;
  for (int i = threadIdx.x; i < 16 * NN; i += 256) {
    float v = p[i];
    s += v; q += v * v;
  }
  ssum[threadIdx.x] = s; ssq[threadIdx.x] = q;
  __syncthreads();
  for (int st = 128; st > 0; st >>= 1) {
    if ((int)threadIdx.x < st) {
      ssum[threadIdx.x] += ssum[threadIdx.x + st];
      ssq[threadIdx.x] += ssq[threadIdx.x + st];
    }
    __syncthreads();
  }
  if (threadIdx.x == 0) {
    const float inv = 1.0f / (16.0f * NN);
    float mu = ssum[0] * inv;
    float var = ssq[0] * inv - mu * mu;
    stats[bg * 2 + 0] = mu;
    stats[bg * 2 + 1] = rsqrtf(var + GN_EPS);
  }
}

// ---------------------------------------------------------------------------
// Kernel 6: apply GroupNorm affine + SiLU
// ---------------------------------------------------------------------------
__global__ void k_gnapply(const float* __restrict__ y, const float* __restrict__ stats,
                          const float* __restrict__ gamma, const float* __restrict__ beta,
                          float* __restrict__ out) {
  size_t i = (size_t)blockIdx.x * 256 + threadIdx.x;  // < B*C*N = 1048576
  int c = (int)((i >> 12) & (CC - 1));
  int bg = (int)(i >> 16);                            // group block = 16*4096
  float mu = stats[bg * 2 + 0];
  float rs = stats[bg * 2 + 1];
  float v = (y[i] - mu) * rs * gamma[c] + beta[c];
  out[i] = v / (1.0f + __expf(-v));
}

// ---------------------------------------------------------------------------
// Launcher
// ---------------------------------------------------------------------------
extern "C" void kernel_launch(void* const* d_in, const int* in_sizes, int n_in,
                              void* d_out, int out_size, void* d_ws, size_t ws_size,
                              hipStream_t stream) {
  const float* x     = (const float*)d_in[0];
  const float* Wq    = (const float*)d_in[1];
  const float* Wk    = (const float*)d_in[2];
  const float* Wv    = (const float*)d_in[3];
  const float* Wo    = (const float*)d_in[4];
  const float* gamma = (const float*)d_in[5];
  const float* beta  = (const float*)d_in[6];
  float* out = (float*)d_out;

  // workspace carve-up (all offsets 256B-aligned)
  char* ws = (char*)d_ws;
  const size_t SZ_BF = (size_t)BB * NN * CC * sizeof(__bf16);   // 2 MB each
  __bf16* xt  = (__bf16*)(ws);                                  // [b][n][c]
  __bf16* wbf = (__bf16*)(ws + SZ_BF);                          // 4x[128][128]
  __bf16* qt  = (__bf16*)(ws + SZ_BF + 131072);                 // [b][n][c]
  __bf16* kt  = (__bf16*)(ws + 2 * SZ_BF + 131072);             // [b][m][c]
  __bf16* vb  = (__bf16*)(ws + 3 * SZ_BF + 131072);             // [b][c][m]
  __bf16* ht  = (__bf16*)(ws + 4 * SZ_BF + 131072);             // [b][n][c]
  float*  yb  = (float*)(ws + 5 * SZ_BF + 131072);              // [b][c][n] f32
  float*  st  = (float*)(ws + 5 * SZ_BF + 131072 +
                         (size_t)BB * CC * NN * sizeof(float)); // 16x{mu,rstd}

  k_transpose<<<dim3(NN / 16, CC / 16, BB), dim3(16, 16), 0, stream>>>(x, xt);
  k_wconv<<<256, 256, 0, stream>>>(Wq, Wk, Wv, Wo, wbf);
  k_proj<<<3072, 128, 0, stream>>>(xt, wbf, qt, kt, vb);   // 12288 waves
  k_attn<<<128, 128, 0, stream>>>(qt, kt, vb, ht);         // 512 waves, 4/block
  k_oproj<<<1024, 128, 0, stream>>>(wbf, ht, x, yb);       // 4096 waves
  k_gnstats<<<BB * NGRP, 256, 0, stream>>>(yb, st);
  k_gnapply<<<(BB * CC * NN) / 256, 256, 0, stream>>>(yb, st, gamma, beta, out);
}
